// TropiGAT_small_module_22351009808613
// MI455X (gfx1250) — matrix operand order repaired and checked by reading, not compiled, not run on hardware
//
#include <hip/hip_runtime.h>
#include <hip/hip_bf16.h>
#include <math.h>

// ---------------- problem constants (match reference) ----------------
#define N_SRC 50000
#define N_DST 20000
#define N_E   640000
#define D_IN  1280
#define NH    4
#define NC    64
#define HC    256        // NH*NC
#define D1    1280
#define D2    480
#define BN_EPS 1e-5f

typedef __attribute__((ext_vector_type(16))) __bf16 v16bf;
typedef __attribute__((ext_vector_type(8)))  float  v8f;

// ---------------- helpers ----------------
__device__ __forceinline__ unsigned enc_f32(float f) {
    unsigned u = __float_as_uint(f);
    return (u & 0x80000000u) ? ~u : (u | 0x80000000u);   // monotone order-preserving
}
__device__ __forceinline__ float dec_f32(unsigned u) {
    unsigned b = (u & 0x80000000u) ? (u ^ 0x80000000u) : ~u;
    return __uint_as_float(b);
}
#define ENC_NEG_INF 0x007FFFFFu   // enc_f32(-inf)

// ---------------- CDNA5 Tensor Data Mover ----------------
#if __has_builtin(__builtin_amdgcn_tensor_load_to_lds)
#define USE_TDM 1
#else
#define USE_TDM 0
#endif

#if USE_TDM
typedef __attribute__((ext_vector_type(4))) unsigned tdm_u4;
typedef __attribute__((ext_vector_type(8))) int      tdm_i8;
typedef __attribute__((ext_vector_type(4))) int      tdm_i4;

// 2D bf16 tile load: LDS gets tile_d1 rows of tile_d0 contiguous elements.
// tensor_d0/d1 = remaining extent from the tile origin (OOB reads return 0).
__device__ __forceinline__ void tdm_load_2d_bf16(
        unsigned lds_off, const __bf16* gptr,
        unsigned tensor_d0, unsigned tensor_d1,
        unsigned tile_d0, unsigned tile_d1,
        unsigned stride_d0_elems) {
    unsigned long long ga = (unsigned long long)(uintptr_t)gptr;
    tdm_u4 g0;
    g0[0] = 1u;                                           // count=1, user descriptor
    g0[1] = lds_off;                                      // LDS byte address
    g0[2] = (unsigned)(ga & 0xFFFFFFFFu);                 // global_addr[31:0]
    g0[3] = (unsigned)((ga >> 32) & 0x01FFFFFFu)          // global_addr[56:32]
          | (2u << 30);                                   // type=2 ("image")
    tdm_i8 g1;
    g1[0] = (int)(1u << 16);                              // wg_mask=0, data_size=1 (2 bytes)
    g1[1] = (int)((tensor_d0 & 0xFFFFu) << 16);           // tensor_dim0 lo16
    g1[2] = (int)(((tensor_d0 >> 16) & 0xFFFFu)
          | ((tensor_d1 & 0xFFFFu) << 16));               // tensor_dim0 hi / dim1 lo
    g1[3] = (int)(((tensor_d1 >> 16) & 0xFFFFu)
          | ((tile_d0 & 0xFFFFu) << 16));                 // tensor_dim1 hi / tile_dim0
    g1[4] = (int)(tile_d1 & 0xFFFFu);                     // tile_dim1, tile_dim2=0
    g1[5] = (int)stride_d0_elems;                         // tensor_dim0_stride[31:0]
    g1[6] = 0;
    g1[7] = 0;
    tdm_i4 z4 = {0, 0, 0, 0};
#if defined(__clang_major__) && (__clang_major__ >= 23)
    tdm_i8 z8 = {0, 0, 0, 0, 0, 0, 0, 0};
    __builtin_amdgcn_tensor_load_to_lds(g0, g1, z4, z4, z8, 0);
#else
    __builtin_amdgcn_tensor_load_to_lds(g0, g1, z4, z4, 0);
#endif
}
#endif // USE_TDM

// ---------------- CDNA5 async global->LDS (ASYNCcnt tracked) ----------------
#if __has_builtin(__builtin_amdgcn_global_load_async_to_lds_b128)
#define USE_ASYNC 1
#else
#define USE_ASYNC 0
#endif

#if USE_ASYNC
// Exact parameter types per hipcc diagnostic: pointer to GCC-style int4 vector,
// global (AS1) source and shared (AS3) destination.
typedef int async_v4i __attribute__((vector_size(16)));
typedef __attribute__((address_space(1))) async_v4i* async_gp128;
typedef __attribute__((address_space(3))) async_v4i* async_lp128;

__device__ __forceinline__ void async_g2l_b128(const void* g, void* l) {
    __builtin_amdgcn_global_load_async_to_lds_b128(
        (async_gp128)(uintptr_t)g,
        (async_lp128)(unsigned)(uintptr_t)l, 0, 0);
}
__device__ __forceinline__ void wait_asynccnt0() {
#if __has_builtin(__builtin_amdgcn_s_wait_asynccnt)
    __builtin_amdgcn_s_wait_asynccnt(0);
#else
    asm volatile("s_wait_asynccnt 0x0" ::: "memory");
#endif
}
#endif // USE_ASYNC

// ---------------- tiny elementwise kernels ----------------
// transposed weight convert: d[n*K + k] = (bf16) s[k*N + n]
__global__ void cvt_transpose_bf16_kernel(const float* __restrict__ s,
                                          __bf16* __restrict__ d, int K, int N) {
    int idx = blockIdx.x * blockDim.x + threadIdx.x;
    if (idx >= K * N) return;
    int k = idx / N, n = idx - k * N;
    d[(size_t)n * K + k] = (__bf16)s[idx];
}

__global__ void init_seg_kernel(unsigned* __restrict__ menc, float* __restrict__ denom, int n) {
    int i = blockIdx.x * blockDim.x + threadIdx.x;
    if (i < n) { menc[i] = ENC_NEG_INF; denom[i] = 0.0f; }
}

__global__ void init_agg_kernel(float* __restrict__ agg, const float* __restrict__ bias_gat, int n) {
    int i = blockIdx.x * blockDim.x + threadIdx.x;
    if (i < n) agg[i] = bias_gat[i & (HC - 1)];
}

// ---------------- BF16 WMMA GEMM: C = epi(A[M,K] @ Bt[N,K]^T) ----------------
// Block: 256 threads = 8 waves.  Block tile 128(M) x 128(N), K step 32.
// Waves: waveM(0..3) x waveN(0..1); each wave = 32(M) x 64(N):
//   two A fragments x four B fragments = EIGHT v_wmma_f32_16x16x32_bf16.
// A: fp32, moved to LDS by GLOBAL_LOAD_ASYNC_TO_LDS_B128, bf16-converted at
//    fragment build.  Bt: bf16 [N][K] pre-transposed weights via TDM.
__device__ __forceinline__ float gemm_epilogue(float x, int n, int mode,
        const float* __restrict__ bias,
        const float* __restrict__ g, const float* __restrict__ be,
        const float* __restrict__ mu, const float* __restrict__ va) {
    x += bias[n];
    if (mode) {
        x = (x - mu[n]) * (g[n] * rsqrtf(va[n] + BN_EPS)) + be[n];
        x = x > 0.0f ? x : 0.01f * x;       // LeakyReLU(0.01)
    }
    return x;
}

__global__ __launch_bounds__(256) void gemm_bf16_wmma_kernel(
        const float* __restrict__ A, const __bf16* __restrict__ Bt,
        float* __restrict__ C,
        const float* __restrict__ bias,
        const float* __restrict__ g, const float* __restrict__ be,
        const float* __restrict__ mu, const float* __restrict__ va,
        int M, int N, int K, int mode) {
    __shared__ float  Asf[128][32];   // [m][k] fp32 tile (async DMA target)
    __shared__ __bf16 Bs[128][32];    // [n][k] bf16, pairs contiguous in k (TDM target)

    const int tid   = threadIdx.x;
    const int lane  = tid & 31;
    const int wid   = tid >> 5;
    const int waveM = wid >> 1;           // 0..3 -> 32 rows each
    const int waveN = wid & 1;            // 0..1 -> 64 cols each
    const int blockM = blockIdx.y * 128;
    const int blockN = blockIdx.x * 128;

    v8f acc[8];
    #pragma unroll
    for (int i = 0; i < 8; ++i) acc[i] = (v8f){};

#if USE_TDM
    const unsigned ldsB = (unsigned)(uintptr_t)(&Bs[0][0]);
#endif

    const int kTiles = K >> 5;
    for (int kt = 0; kt < kTiles; ++kt) {
        const int k0 = kt << 5;
        __syncthreads();                         // previous tile fully consumed
        // --- issue B: 128(n) x 32(k) bf16 via Tensor Data Mover (wave 0) ---
#if USE_TDM
        if (wid == 0) {
            tdm_load_2d_bf16(ldsB, Bt + (size_t)blockN * K + k0,
                             (unsigned)(K - k0), (unsigned)(N - blockN),
                             32u, 128u, (unsigned)K);
        }
#else
        for (int i = tid; i < 128 * 4; i += 256) {        // 4x uint4 per row
            int n = i >> 2, q = i & 3;
            uint4 v = make_uint4(0u, 0u, 0u, 0u);
            if (blockN + n < N)
                v = *(const uint4*)(Bt + (size_t)(blockN + n) * K + k0 + q * 8);
            *(uint4*)&Bs[n][q * 8] = v;
        }
#endif
        // --- issue A: 128 x 32 fp32, 16B per lane per iteration (overlaps TDM) ---
        for (int i = tid; i < 128 * 8; i += 256) {
            int r = i >> 3, q = i & 7;
            int gm = blockM + r;
            if (gm >= M) gm = M - 1;              // clamp; stores are guarded
            const float* src = A + (size_t)gm * K + k0 + q * 4;
#if USE_ASYNC
            async_g2l_b128(src, &Asf[r][q * 4]);
#else
            *(float4*)&Asf[r][q * 4] = *(const float4*)src;
#endif
        }
#if USE_ASYNC
        wait_asynccnt0();
#endif
#if USE_TDM
        if (wid == 0) __builtin_amdgcn_s_wait_tensorcnt(0);
#endif
        __syncthreads();

        // --- build fragments (ISA 7.12.2 layouts) ---
        union Frag { v16bf v; __bf16 h[16]; unsigned u[8]; };
        Frag af[2], bfr[4];
        const int apb = (lane >= 16) ? 4 : 0;     // fp32-pair base: K 8..15 for hi lanes
        #pragma unroll
        for (int s = 0; s < 2; ++s) {
            const int am = waveM * 32 + s * 16 + (lane & 15);
            const float2* arow = (const float2*)&Asf[am][0];   // 16 K-pairs
            #pragma unroll
            for (int j = 0; j < 4; ++j) {
                float2 p0 = arow[apb + j];        // K {0..7} | {8..15}
                float2 p1 = arow[8 + apb + j];    // K {16..23} | {24..31}
                af[s].h[2 * j]     = (__bf16)p0.x;
                af[s].h[2 * j + 1] = (__bf16)p0.y;
                af[s].h[8 + 2 * j] = (__bf16)p1.x;
                af[s].h[9 + 2 * j] = (__bf16)p1.y;
            }
        }
        const int bpb = (lane >= 16) ? 4 : 0;     // bf16-pair base: K 8..15 | 24..31 for hi lanes
        #pragma unroll
        for (int t = 0; t < 4; ++t) {
            const int nl = waveN * 64 + t * 16 + (lane & 15);
            const unsigned* brow = (const unsigned*)&Bs[nl][0];
            #pragma unroll
            for (int j = 0; j < 8; ++j) bfr[t].u[j] = brow[bpb + (j & 3) + ((j >> 2) << 3)];
        }
        #pragma unroll
        for (int s = 0; s < 2; ++s)
            #pragma unroll
            for (int t = 0; t < 4; ++t)
                acc[s * 4 + t] = __builtin_amdgcn_wmma_f32_16x16x32_bf16(
                        false, af[s].v, false, bfr[t].v, (short)0, acc[s * 4 + t],
                        false, false);
    }

    // --- epilogue: C/D layout -> VGPR v holds row (v + 8*hi), lane&15 = col ---
    #pragma unroll
    for (int s = 0; s < 2; ++s) {
        const int mBase = blockM + waveM * 32 + s * 16 + ((lane >= 16) ? 8 : 0);
        #pragma unroll
        for (int t = 0; t < 4; ++t) {
            const int n = blockN + waveN * 64 + t * 16 + (lane & 15);
            if (n < N) {
                #pragma unroll
                for (int v = 0; v < 8; ++v) {
                    int gm = mBase + v;
                    if (gm < M)
                        C[(size_t)gm * N + n] =
                            gemm_epilogue(acc[s * 4 + t][v], n, mode, bias, g, be, mu, va);
                }
            }
        }
    }
}

// ---------------- edge pass 1: logits + segment max ----------------
__global__ void edge_logits_kernel(const float* __restrict__ xl, const float* __restrict__ xr,
                                   const int* __restrict__ src, const int* __restrict__ dst,
                                   const float* __restrict__ att,
                                   float* __restrict__ e, unsigned* __restrict__ menc) {
    int idx = blockIdx.x * blockDim.x + threadIdx.x;      // (edge, head)
    if (idx >= N_E * NH) return;
    int edge = idx >> 2, h = idx & 3;
    int s = src[edge], d = dst[edge];
    const float4* l = (const float4*)(xl + (size_t)s * HC + h * NC);
    const float4* r = (const float4*)(xr + (size_t)d * HC + h * NC);
    const float4* a = (const float4*)(att + h * NC);
    float sum = 0.0f;
    #pragma unroll
    for (int j = 0; j < NC / 4; ++j) {
        float4 lv = l[j], rv = r[j], av = a[j];
        float t;
        t = lv.x + rv.x; t = t > 0.f ? t : 0.2f * t; sum += t * av.x;
        t = lv.y + rv.y; t = t > 0.f ? t : 0.2f * t; sum += t * av.y;
        t = lv.z + rv.z; t = t > 0.f ? t : 0.2f * t; sum += t * av.z;
        t = lv.w + rv.w; t = t > 0.f ? t : 0.2f * t; sum += t * av.w;
    }
    e[idx] = sum;
    atomicMax(menc + (size_t)d * NH + h, enc_f32(sum));
}

// ---------------- edge pass 2: exp + segment sum ----------------
__global__ void edge_exp_kernel(const int* __restrict__ dst,
                                const float* __restrict__ e, const unsigned* __restrict__ menc,
                                float* __restrict__ ex, float* __restrict__ denom) {
    int idx = blockIdx.x * blockDim.x + threadIdx.x;
    if (idx >= N_E * NH) return;
    int edge = idx >> 2, h = idx & 3;
    int d = dst[edge];
    float m = dec_f32(menc[(size_t)d * NH + h]);
    if (!isfinite(m)) m = 0.0f;                          // matches reference
    float v = __expf(e[idx] - m);
    ex[idx] = v;
    atomicAdd(denom + (size_t)d * NH + h, v);
}

// ---------------- edge pass 3: normalize + scatter aggregate ----------------
__global__ __launch_bounds__(256) void edge_agg_kernel(
        const float* __restrict__ xl,
        const int* __restrict__ src, const int* __restrict__ dst,
        const float* __restrict__ ex, const float* __restrict__ denom,
        float* __restrict__ agg) {
    int edge = blockIdx.x;
    int c = threadIdx.x;                 // 0..255
    int h = c >> 6;
    int s = src[edge], d = dst[edge];
    float dn = denom[(size_t)d * NH + h];
    dn = dn > 0.0f ? dn : 1.0f;
    float alpha = ex[(size_t)edge * NH + h] / dn;
    atomicAdd(agg + (size_t)d * HC + c, xl[(size_t)s * HC + c] * alpha);
}

// ---------------- final: y = h2 @ W3 + b3 ----------------
__global__ __launch_bounds__(128) void final_dot_kernel(const float* __restrict__ h2,
                                                        const float* __restrict__ W3,
                                                        const float* __restrict__ b3,
                                                        float* __restrict__ y) {
    __shared__ float red[128];
    int row = blockIdx.x;
    int t = threadIdx.x;
    float p = 0.0f;
    for (int k = t; k < D2; k += 128) p += h2[(size_t)row * D2 + k] * W3[k];
    red[t] = p;
    __syncthreads();
    for (int off = 64; off > 0; off >>= 1) {
        if (t < off) red[t] += red[t + off];
        __syncthreads();
    }
    if (t == 0) y[row] = red[0] + b3[0];
}

// ---------------- host launcher ----------------
static inline size_t alignup(size_t x) { return (x + 255) & ~(size_t)255; }

extern "C" void kernel_launch(void* const* d_in, const int* in_sizes, int n_in,
                              void* d_out, int out_size, void* d_ws, size_t ws_size,
                              hipStream_t stream) {
    const float* x_src   = (const float*)d_in[0];
    const float* x_dst   = (const float*)d_in[1];
    const int*   src_idx = (const int*)d_in[2];
    const int*   dst_idx = (const int*)d_in[3];
    const float* W       = (const float*)d_in[4];
    const float* b_lin   = (const float*)d_in[5];
    const float* att     = (const float*)d_in[6];
    const float* bias_gat= (const float*)d_in[7];
    const float* W1      = (const float*)d_in[8];
    const float* b1      = (const float*)d_in[9];
    const float* bn1_g   = (const float*)d_in[10];
    const float* bn1_b   = (const float*)d_in[11];
    const float* bn1_m   = (const float*)d_in[12];
    const float* bn1_v   = (const float*)d_in[13];
    const float* W2      = (const float*)d_in[14];
    const float* b2      = (const float*)d_in[15];
    const float* bn2_g   = (const float*)d_in[16];
    const float* bn2_b   = (const float*)d_in[17];
    const float* bn2_m   = (const float*)d_in[18];
    const float* bn2_v   = (const float*)d_in[19];
    const float* W3      = (const float*)d_in[20];
    const float* b3      = (const float*)d_in[21];
    float* y = (float*)d_out;

    // workspace layout (transposed bf16 weights: Bt[N][K])
    char* p = (char*)d_ws;
    size_t off = 0;
    __bf16* Wt  = (__bf16*)(p + off); off = alignup(off + (size_t)HC * D_IN * 2);
    __bf16* W1t = (__bf16*)(p + off); off = alignup(off + (size_t)D1 * HC * 2);
    __bf16* W2t = (__bf16*)(p + off); off = alignup(off + (size_t)D2 * D1 * 2);
    float* xl    = (float*)(p + off); off = alignup(off + (size_t)N_SRC * HC * 4);
    float* xr    = (float*)(p + off); off = alignup(off + (size_t)N_DST * HC * 4);
    float* e     = (float*)(p + off); off = alignup(off + (size_t)N_E * NH * 4);
    float* ex    = (float*)(p + off); off = alignup(off + (size_t)N_E * NH * 4);
    unsigned* mE = (unsigned*)(p + off); off = alignup(off + (size_t)N_DST * NH * 4);
    float* denom = (float*)(p + off); off = alignup(off + (size_t)N_DST * NH * 4);
    float* agg   = (float*)(p + off); off = alignup(off + (size_t)N_DST * HC * 4);
    float* h1    = (float*)(p + off); off = alignup(off + (size_t)N_DST * D1 * 4);
    float* h2    = (float*)(p + off); off = alignup(off + (size_t)N_DST * D2 * 4);
    (void)ws_size; (void)in_sizes; (void)n_in; (void)out_size;

    // 1) weight conversions to transposed bf16 (TDM-friendly [N][K] row-major)
    {
        int n = D_IN * HC;
        cvt_transpose_bf16_kernel<<<(n + 255) / 256, 256, 0, stream>>>(W, Wt, D_IN, HC);
        n = HC * D1;
        cvt_transpose_bf16_kernel<<<(n + 255) / 256, 256, 0, stream>>>(W1, W1t, HC, D1);
        n = D1 * D2;
        cvt_transpose_bf16_kernel<<<(n + 255) / 256, 256, 0, stream>>>(W2, W2t, D1, D2);
    }
    // 2) segment state init
    {
        int n = N_DST * NH;
        init_seg_kernel<<<(n + 255) / 256, 256, 0, stream>>>(mE, denom, n);
        n = N_DST * HC;
        init_agg_kernel<<<(n + 255) / 256, 256, 0, stream>>>(agg, bias_gat, n);
    }
    // 3) shared linear: xl = x_src@W + b_lin ; xr = x_dst@W + b_lin
    {
        dim3 blk(256);
        dim3 gs((HC + 127) / 128, (N_SRC + 127) / 128);
        gemm_bf16_wmma_kernel<<<gs, blk, 0, stream>>>(x_src, Wt, xl, b_lin,
                nullptr, nullptr, nullptr, nullptr, N_SRC, HC, D_IN, 0);
        dim3 gd((HC + 127) / 128, (N_DST + 127) / 128);
        gemm_bf16_wmma_kernel<<<gd, blk, 0, stream>>>(x_dst, Wt, xr, b_lin,
                nullptr, nullptr, nullptr, nullptr, N_DST, HC, D_IN, 0);
    }
    // 4) edge passes
    {
        int n = N_E * NH;
        edge_logits_kernel<<<(n + 255) / 256, 256, 0, stream>>>(xl, xr, src_idx, dst_idx, att, e, mE);
        edge_exp_kernel<<<(n + 255) / 256, 256, 0, stream>>>(dst_idx, e, mE, ex, denom);
        edge_agg_kernel<<<N_E, 256, 0, stream>>>(xl, src_idx, dst_idx, ex, denom, agg);
    }
    // 5) MLP head
    {
        dim3 blk(256);
        dim3 g1((D1 + 127) / 128, (N_DST + 127) / 128);
        gemm_bf16_wmma_kernel<<<g1, blk, 0, stream>>>(agg, W1t, h1, b1,
                bn1_g, bn1_b, bn1_m, bn1_v, N_DST, D1, HC, 1);
        dim3 g2((D2 + 127) / 128, (N_DST + 127) / 128);
        gemm_bf16_wmma_kernel<<<g2, blk, 0, stream>>>(h1, W2t, h2, b2,
                bn2_g, bn2_b, bn2_m, bn2_v, N_DST, D2, D1, 1);
        final_dot_kernel<<<N_DST, 128, 0, stream>>>(h2, W3, b3, y);
    }
}
